// SupConLoss_73005854097688
// MI455X (gfx1250) — matrix operand order repaired and checked by reading, not compiled, not run on hardware
//
#include <hip/hip_runtime.h>
#include <hip/hip_bf16.h>
#include <math.h>

typedef __attribute__((ext_vector_type(16))) __bf16 v16bf;
typedef __attribute__((ext_vector_type(8)))  __bf16 v8bf;
typedef __attribute__((ext_vector_type(8)))  float  v8f;

#define WG 256
#define DIM 128          // feature dimension (K of the matmul)
#define ROWS_PER_WG 128  // 8 waves x 16 rows
#define MAX_B 8192

// ---------------------------------------------------------------------------
// Kernel 1: scale features by sqrt(log2(e)/T) and split fp32 -> bf16 hi + lo.
// sim in exp2 domain: y_ij = (g_i . g_j), g = f * sqrt(log2e/5).
// ---------------------------------------------------------------------------
__global__ __launch_bounds__(WG) void supcon_split(const float* __restrict__ f,
                                                   __bf16* __restrict__ hi,
                                                   __bf16* __restrict__ lo,
                                                   int n, float scale) {
  int i = blockIdx.x * blockDim.x + threadIdx.x;
  if (i < n) {
    float g = f[i] * scale;
    __bf16 h = (__bf16)g;
    hi[i] = h;
    lo[i] = (__bf16)(g - (float)h);
  }
}

// ---------------------------------------------------------------------------
// Fragment loaders (CDNA5 16-bit WMMA VGPR layouts, CDNA5 ISA 7.12.2)
// A 16x32:  lane = {m=lane&15, khalf=lane>>4}; elems 0..7 -> k=khalf*8+0..7,
//           elems 8..15 -> k=16+khalf*8+0..7  (two aligned 16B loads)
// B 32x16:  lane = {n=lane&15, kgrp=lane>>4};  elems 0..7 -> k=kgrp*8+0..7, 8..15 -> k=16+kgrp*8+0..7
//           (two aligned 16B loads)
// ---------------------------------------------------------------------------
__device__ inline v16bf load_a_frag(const __bf16* __restrict__ base, int row,
                                    int kbase, int akh) {
  const __bf16* p = base + (size_t)row * DIM + kbase + akh * 8;
  v8bf x0 = *(const v8bf*)(p);
  v8bf x1 = *(const v8bf*)(p + 16);
  v16bf r;
#pragma unroll
  for (int e = 0; e < 8; ++e) { r[e] = x0[e]; r[e + 8] = x1[e]; }
  return r;
}

__device__ inline v16bf load_b_frag(const __bf16* __restrict__ base, int col,
                                    int kbase, int bkg) {
  const __bf16* p = base + (size_t)col * DIM + kbase + bkg * 8;
  v8bf x0 = *(const v8bf*)(p);
  v8bf x1 = *(const v8bf*)(p + 16);
  v16bf r;
#pragma unroll
  for (int e = 0; e < 8; ++e) { r[e] = x0[e]; r[e + 8] = x1[e]; }
  return r;
}

// ---------------------------------------------------------------------------
// Kernel 2: fused sim + softmax stats. One wave owns 16 rows, streams all
// columns in 16-wide tiles. Unshifted exp2 accumulation (exact: the max shift
// is a power-of-two scale; off-diag |y| <= ~20 so no overflow), max tracked
// separately, diagonal tile peeled. Emits per-row mean_log_prob_pos * recip.
// ---------------------------------------------------------------------------
__global__ __launch_bounds__(WG) void supcon_main(const __bf16* __restrict__ Ghi,
                                                  const __bf16* __restrict__ Glo,
                                                  const int* __restrict__ labels,
                                                  float* __restrict__ row_out,
                                                  int B) {
  __shared__ int slab[MAX_B];
  for (int i = threadIdx.x; i < B; i += WG) slab[i] = labels[i];
  __syncthreads();

  const int wave = threadIdx.x >> 5;
  const int lane = threadIdx.x & 31;
  const int am   = lane & 15;   // A: row-in-strip / B: col-in-tile
  const int akh  = lane >> 4;   // A: k-half / B: k-group
  const int r0   = blockIdx.x * ROWS_PER_WG + wave * 16;

  // Resident A fragments: rows r0..r0+15, K = 4 chunks of 32, hi & lo parts.
  v16bf ahi[4], alo[4];
#pragma unroll
  for (int kc = 0; kc < 4; ++kc) {
    ahi[kc] = load_a_frag(Ghi, r0 + am, kc * 32, akh);
    alo[kc] = load_a_frag(Glo, r0 + am, kc * 32, akh);
  }

  // Per-lane stats for 8 rows (C layout: VGPR v -> M = v + 8*(lane>=16)).
  float m[8], s[8], p[8];
  int cnt[8], rl[8];
#pragma unroll
  for (int v = 0; v < 8; ++v) {
    m[v] = -1.0e30f; s[v] = 0.0f; p[v] = 0.0f; cnt[v] = 0;
    rl[v] = slab[r0 + v + 8 * akh];
  }

  for (int c0 = 0; c0 < B; c0 += 16) {
    // Three independent WMMA accumulator chains: hi*hi, hi*lo, lo*hi.
    v8f a0 = {}, a1 = {}, a2 = {};
#pragma unroll
    for (int kc = 0; kc < 4; ++kc) {
      v16bf bh = load_b_frag(Ghi, c0 + am, kc * 32, akh);
      v16bf bl = load_b_frag(Glo, c0 + am, kc * 32, akh);
      a0 = __builtin_amdgcn_wmma_f32_16x16x32_bf16(false, ahi[kc], false, bh,
                                                   (short)0, a0, false, false);
      a1 = __builtin_amdgcn_wmma_f32_16x16x32_bf16(false, ahi[kc], false, bl,
                                                   (short)0, a1, false, false);
      a2 = __builtin_amdgcn_wmma_f32_16x16x32_bf16(false, alo[kc], false, bh,
                                                   (short)0, a2, false, false);
    }

    const int cl = slab[c0 + am];  // this lane's column label

    if (c0 == r0) {
      // The single tile containing this strip's diagonal (wave-uniform branch).
#pragma unroll
      for (int v = 0; v < 8; ++v) {
        float y    = (a0[v] + a1[v]) + a2[v];
        bool  diag = (am == v + 8 * akh);
        bool  pos  = (!diag) && (cl == rl[v]);
        m[v]    = fmaxf(m[v], y);                  // row max INCLUDES diagonal
        s[v]   += diag ? 0.0f : __builtin_amdgcn_exp2f(y);
        p[v]   += pos ? y : 0.0f;
        cnt[v] += pos ? 1 : 0;
      }
    } else {
      // Fast path: no diagonal in this tile.
#pragma unroll
      for (int v = 0; v < 8; ++v) {
        float y   = (a0[v] + a1[v]) + a2[v];
        bool  pos = (cl == rl[v]);
        m[v]    = fmaxf(m[v], y);
        s[v]   += __builtin_amdgcn_exp2f(y);
        p[v]   += pos ? y : 0.0f;
        cnt[v] += pos ? 1 : 0;
      }
    }
  }

  // Merge the 16 lanes holding each row (xor over bits 0..3 stays in half).
  // Unshifted sums merge with plain adds; max with fmax.
#pragma unroll
  for (int off = 1; off <= 8; off <<= 1) {
#pragma unroll
    for (int v = 0; v < 8; ++v) {
      m[v]    = fmaxf(m[v], __shfl_xor(m[v], off, 32));
      s[v]   += __shfl_xor(s[v], off, 32);
      p[v]   += __shfl_xor(p[v], off, 32);
      cnt[v] += __shfl_xor(cnt[v], off, 32);
    }
  }

  if (am == 0) {  // lanes 0 and 16 publish rows v and v+8
    const float LN2 = 0.6931471805599453f;
#pragma unroll
    for (int v = 0; v < 8; ++v) {
      float fc = (float)cnt[v];
      // Recover the reference's shifted sum exactly: S = s * 2^(-rowmax).
      float S  = s[v] * __builtin_amdgcn_exp2f(-m[v]);
      float logden     = __builtin_amdgcn_logf(1e-12f + S) * LN2;  // natural log
      float sum_logits = (p[v] - fc * m[v]) * LN2;                 // exp2 -> e dom
      float mlpp       = sum_logits - fc * logden;
      float recip      = cnt[v] > 0 ? 1.0f / fc : 0.0f;
      row_out[r0 + v + 8 * akh] = mlpp * recip;
    }
  }
}

// ---------------------------------------------------------------------------
// Kernel 3: deterministic fixed-order reduction -> loss = -mean(row_vals)
// ---------------------------------------------------------------------------
__global__ __launch_bounds__(WG) void supcon_reduce(const float* __restrict__ rows,
                                                    float* __restrict__ out, int B) {
  __shared__ float sm[WG];
  float acc = 0.0f;
  for (int i = threadIdx.x; i < B; i += WG) acc += rows[i];
  sm[threadIdx.x] = acc;
  __syncthreads();
  for (int st = WG / 2; st > 0; st >>= 1) {
    if (threadIdx.x < st) sm[threadIdx.x] += sm[threadIdx.x + st];
    __syncthreads();
  }
  if (threadIdx.x == 0) out[0] = -sm[0] / (float)B;
}

// ---------------------------------------------------------------------------
extern "C" void kernel_launch(void* const* d_in, const int* in_sizes, int n_in,
                              void* d_out, int out_size, void* d_ws, size_t ws_size,
                              hipStream_t stream) {
  const float* feats  = (const float*)d_in[0];
  const int*   labels = (const int*)d_in[1];
  // d_in[2] = fac_label: has no effect in the reference (mask_cross == ones)

  const int nfeat = in_sizes[0];          // B * D
  const int B     = in_sizes[1];          // 8192
  float* out = (float*)d_out;

  char* ws = (char*)d_ws;
  __bf16* Ghi   = (__bf16*)ws;                                   // nfeat bf16
  __bf16* Glo   = (__bf16*)(ws + (size_t)nfeat * sizeof(__bf16));
  float*  rowsv = (float*)(ws + 2 * (size_t)nfeat * sizeof(__bf16));

  // Fold temperature + exp2 conversion symmetrically into both operands.
  const float scale = sqrtf(1.4426950408889634f / 5.0f);

  supcon_split<<<(nfeat + WG - 1) / WG, WG, 0, stream>>>(feats, Ghi, Glo, nfeat, scale);
  supcon_main<<<B / ROWS_PER_WG, WG, 0, stream>>>(Ghi, Glo, labels, rowsv, B);
  supcon_reduce<<<1, WG, 0, stream>>>(rowsv, out, B);
}
